// MultiHeadsAttention_89584427860485
// MI455X (gfx1250) — matrix operand order repaired and checked by reading, not compiled, not run on hardware
//
#include <hip/hip_runtime.h>
#include <hip/hip_bf16.h>

typedef __attribute__((ext_vector_type(16))) _Float16 v16h;
typedef __attribute__((ext_vector_type(8)))  _Float16 v8h;
typedef __attribute__((ext_vector_type(4)))  _Float16 v4h;
typedef __attribute__((ext_vector_type(8)))  float    v8f;

#define WMMA_F16(a, b, c) \
    __builtin_amdgcn_wmma_f32_16x16x32_f16(false, (a), false, (b), (short)0, (c), false, false)

// Problem dims
constexpr int BB = 4, SS = 2048, DD = 1024, HH = 16, HD = 64;
constexpr int MROWS = BB * SS;   // 8192 tokens

// Workspace layout (bytes), all 16B aligned
constexpr size_t OFF_XH  = 0;         // f16 X    [8192,1024]   16 MB
constexpr size_t OFF_WTH = 16777216;  // f16 W^T  [3,1024,1024]  6 MB
constexpr size_t OFF_QH  = 23068672;  // f16 Q    [B,H,S,HD]    16 MB
constexpr size_t OFF_KH  = 39845888;  // f16 K    [B,H,S,HD]    16 MB
constexpr size_t OFF_VTH = 56623104;  // f16 V^T  [B,H,HD,S]    16 MB

__device__ __forceinline__ v16h load_frag(const _Float16* p0, const _Float16* p1) {
    union { v16h v; v8h h[2]; } u;
    u.h[0] = *(const v8h*)p0;
    u.h[1] = *(const v8h*)p1;
    return u.v;
}

// ---- CDNA5 async global->LDS copy (ASYNCcnt-tracked), inline asm form -----
// Generic pointers to __shared__ carry the LDS offset in their low 32 bits.
__device__ __forceinline__ void async_ld_b128(void* lds_ptr, const void* gptr) {
    unsigned loff = (unsigned)(unsigned long long)lds_ptr;
    asm volatile("global_load_async_to_lds_b128 %0, %1, off"
                 :: "v"(loff), "v"(gptr)
                 : "memory");
}
__device__ __forceinline__ void wait_async0() {
    asm volatile("s_wait_asynccnt 0" ::: "memory");
}

// ---------------------------------------------------------------------------
// Kernel 0a: X fp32 -> f16, vectorized x4
// ---------------------------------------------------------------------------
__global__ void __launch_bounds__(256)
cvt_x_kernel(const float* __restrict__ X, _Float16* __restrict__ Xh, int n4) {
    int i = blockIdx.x * blockDim.x + threadIdx.x;
    if (i >= n4) return;
    float4 f = ((const float4*)X)[i];
    v4h p = { (_Float16)f.x, (_Float16)f.y, (_Float16)f.z, (_Float16)f.w };
    ((v4h*)Xh)[i] = p;
}

// ---------------------------------------------------------------------------
// Kernel 0b: Wq/Wk/Wv fp32 [K,N] -> f16 transposed [mat][N][K]
// ---------------------------------------------------------------------------
__global__ void __launch_bounds__(256)
cvt_wt_kernel(const float* __restrict__ Wq, const float* __restrict__ Wk,
              const float* __restrict__ Wv, _Float16* __restrict__ Wth) {
    int tid = blockIdx.x * blockDim.x + threadIdx.x;     // 0 .. 3*2^20
    int mat = tid >> 20;
    int rem = tid & 1048575;
    int k = rem >> 10;
    int n = rem & 1023;
    const float* W = (mat == 0) ? Wq : (mat == 1) ? Wk : Wv;
    Wth[(size_t)mat * 1048576 + (size_t)n * 1024 + k] = (_Float16)W[rem];
}

// ---------------------------------------------------------------------------
// Kernel 1: fused QKV projection + bias + RoPE
//   block = 8 waves: 128 token rows x 64 cols; wave tile 16x64 (4 WMMA tiles,
//   32 acc VGPRs). W k-slab (64n x 32k, 4KB) double-buffered in LDS via one
//   async b128 per thread. A fragment software-pipelined one k-step ahead.
// ---------------------------------------------------------------------------
__global__ void __launch_bounds__(256)
proj_rope_kernel(const _Float16* __restrict__ Xh, const _Float16* __restrict__ Wth,
                 const float* __restrict__ bq, const float* __restrict__ bk,
                 const float* __restrict__ bv,
                 _Float16* __restrict__ Qh, _Float16* __restrict__ Kh,
                 _Float16* __restrict__ Vth) {
    __shared__ __align__(16) _Float16 Bt[2][64][32];     // [buf][n][k] 8 KB

    const int tid  = threadIdx.x;
    const int lane = tid & 31;
    const int widx = tid >> 5;
    const int ngrp = blockIdx.x % 48;                    // 48 groups of 64 cols
    const int mgrp = blockIdx.x / 48;                    // 64 groups of 128 rows
    const int nbase   = ngrp * 64;
    const int rowbase = mgrp * 128 + widx * 16;
    const int ln = lane & 15, hl = lane >> 4;

    // staging: 64 rows x 64B = 256 b128 chunks, one per thread
    const int srow = tid >> 2;                           // 0..63
    const int soff = (tid & 3) * 8;                      // elements
    const _Float16* Wrow_g = Wth + (size_t)(nbase + srow) * DD + soff;
    auto stage = [&](int b, int kk) {
        async_ld_b128(&Bt[b][srow][soff], Wrow_g + kk * 32);
    };

    const _Float16* Arow = Xh + (size_t)(rowbase + ln) * DD;

    const v8f vzero = {0.f, 0.f, 0.f, 0.f, 0.f, 0.f, 0.f, 0.f};
    v8f acc[4];
#pragma unroll
    for (int t = 0; t < 4; ++t) acc[t] = vzero;

    stage(0, 0);
    v16h a = load_frag(Arow + hl * 8, Arow + 16 + hl * 8);   // A for kk=0

#pragma unroll 1
    for (int kk = 0; kk < 32; ++kk) {
        wait_async0();            // my slice of current buffer has landed
        __syncthreads();          // all slices landed; prev readers done
        if (kk + 1 < 32) stage((kk + 1) & 1, kk + 1);

        // prefetch next A fragment (global) before this step's WMMAs
        v16h an = a;
        if (kk + 1 < 32) {
            const int kn = (kk + 1) * 32;
            an = load_frag(Arow + kn + hl * 8, Arow + kn + 16 + hl * 8);
        }

        const int buf = kk & 1;
        v16h bf[4];
#pragma unroll
        for (int t = 0; t < 4; ++t) {
            const _Float16* br = &Bt[buf][t * 16 + ln][hl * 8];
            bf[t] = load_frag(br, br + 16);
        }
#pragma unroll
        for (int t = 0; t < 4; ++t)
            acc[t] = WMMA_F16(a, bf[t], acc[t]);

        a = an;
    }

    const int b_    = rowbase >> 11;   // batch (tile never crosses batch)
    const int sbase = rowbase & 2047;  // seq position base

#pragma unroll 1
    for (int t = 0; t < 4; ++t) {
        const int ng  = nbase + t * 16 + ln;     // global fused column
        const int mat = ng >> 10;                // 0=q 1=k 2=v (uniform per wave)
        const int col = ng & 1023;
        const int h = col >> 6, hd = col & 63;
        const float* bp = (mat == 0) ? bq : (mat == 1) ? bk : bv;
        const float bias = bp[col];

        float vals[8];
#pragma unroll
        for (int i = 0; i < 8; ++i) vals[i] = acc[t][i] + bias;

        if (mat < 2) {
            // RoPE: inv_freq = 10000^(-(2*(hd/2))/64); pairs = adjacent lanes
            const float freq = __expf(-(float)(hd & ~1) * (9.210340371976184f / 64.0f));
            const float sgn  = (hd & 1) ? 1.0f : -1.0f;
            _Float16* dst = (mat == 0) ? Qh : Kh;
#pragma unroll
            for (int i = 0; i < 8; ++i) {
                const int sr = sbase + i + 8 * hl;
                float sn, cs;
                __sincosf((float)sr * freq, &sn, &cs);
                const float partner = __shfl_xor(vals[i], 1, 32);
                const float r = vals[i] * cs + sgn * partner * sn;
                dst[((size_t)(b_ * HH + h) * SS + sr) * HD + hd] = (_Float16)r;
            }
        } else {
            v8h pk;
#pragma unroll
            for (int i = 0; i < 8; ++i) pk[i] = (_Float16)vals[i];
            const size_t vb =
                ((size_t)(b_ * HH + h) * HD + hd) * SS + sbase + 8 * hl;
            *(v8h*)(Vth + vb) = pk;
        }
    }
}

// ---------------------------------------------------------------------------
// Kernel 2: flash-style attention.
//   block = 8 waves sharing one (b,h); K tile (32x64) and V^T tile (64x32)
//   double-buffered in LDS via async copies (1 K + 1 V b128 chunk/thread).
//   Each wave: 4 score WMMAs -> raw score writeback -> online softmax ->
//   P transpose via LDS -> 4 PV WMMAs.
// ---------------------------------------------------------------------------
__global__ void __launch_bounds__(256)
attn_kernel(const _Float16* __restrict__ Qh, const _Float16* __restrict__ Kh,
            const _Float16* __restrict__ Vth,
            float* __restrict__ ctx, float* __restrict__ scores) {
    __shared__ __align__(16) _Float16 Kt[2][32][64];     // [buf][kpos][hd]  8 KB
    __shared__ __align__(16) _Float16 Vt[2][64][32];     // [buf][hd][kpos]  8 KB
    __shared__ __align__(64) _Float16 Pbuf[8][16][32];   // per-wave P stage 8 KB

    const int tid  = threadIdx.x;
    const int lane = tid & 31;
    const int widx = tid >> 5;
    const int bh = blockIdx.x >> 4;                      // b*16+h, shared by block
    const int qt = (blockIdx.x & 15) * 8 + widx;         // 128 q-tiles per bh
    const int ln = lane & 15, hl = lane >> 4;
    const int q0 = qt * 16;

    const _Float16* Qb = Qh  + (size_t)bh * SS * HD;
    const _Float16* Kb = Kh  + (size_t)bh * SS * HD;
    const _Float16* Vb = Vth + (size_t)bh * HD * SS;
    float* sc = scores + (size_t)bh * SS * SS;

    // stage K (32 rows x 128B) + V^T (64 rows x 64B): 1+1 b128 chunks/thread
    const int krow = tid >> 3, koff = (tid & 7) * 8;     // K: 8 chunks/row
    const int vrow = tid >> 2, voff = (tid & 3) * 8;     // V: 4 chunks/row
    auto stage = [&](int b, int kt) {
        const int kb = kt * 32;
        async_ld_b128(&Kt[b][krow][koff], Kb + (size_t)(kb + krow) * HD + koff);
        async_ld_b128(&Vt[b][vrow][voff], Vb + (size_t)vrow * SS + kb + voff);
    };

    // Q fragments (16 rows x 64 hd) resident in registers
    const _Float16* qrow = Qb + (size_t)(q0 + ln) * HD;
    const v16h aq0 = load_frag(qrow + hl * 8,      qrow + 16 + hl * 8);
    const v16h aq1 = load_frag(qrow + 32 + hl * 8, qrow + 48 + hl * 8);

    const v8f vzero = {0.f, 0.f, 0.f, 0.f, 0.f, 0.f, 0.f, 0.f};
    v8f acc[4];
#pragma unroll
    for (int d = 0; d < 4; ++d) acc[d] = vzero;
    float mrow[8], lsum[8];
#pragma unroll
    for (int i = 0; i < 8; ++i) { mrow[i] = -INFINITY; lsum[i] = 0.f; }

    stage(0, 0);
#pragma unroll 1
    for (int kt = 0; kt < 64; ++kt) {
        wait_async0();
        __syncthreads();
        if (kt + 1 < 64) stage((kt + 1) & 1, kt + 1);

        const int buf = kt & 1;
        const int kb  = kt * 32;

        // ---- scores: two 16x16 tiles; batch the 4 B fragments ------------
        v16h kf[4];
        {
            const _Float16* kr = &Kt[buf][ln][hl * 8];
            kf[0] = load_frag(kr,      kr + 16);
            kf[1] = load_frag(kr + 32, kr + 48);
            const _Float16* kr1 = &Kt[buf][16 + ln][hl * 8];
            kf[2] = load_frag(kr1,      kr1 + 16);
            kf[3] = load_frag(kr1 + 32, kr1 + 48);
        }
        v8f c0 = vzero, c1 = vzero;
        c0 = WMMA_F16(aq0, kf[0], c0);
        c0 = WMMA_F16(aq1, kf[1], c0);
        c1 = WMMA_F16(aq0, kf[2], c1);
        c1 = WMMA_F16(aq1, kf[3], c1);
#pragma unroll
        for (int i = 0; i < 8; ++i) { c0[i] *= 0.03125f; c1[i] *= 0.03125f; }

        // ---- raw score writeback (reference returns pre-softmax scores) ---
#pragma unroll
        for (int i = 0; i < 8; ++i) {
            const int row = q0 + i + 8 * hl;
            sc[(size_t)row * SS + kb + ln]      = c0[i];
            sc[(size_t)row * SS + kb + 16 + ln] = c1[i];
        }

        // ---- online softmax over this 32-wide chunk ----
#pragma unroll
        for (int i = 0; i < 8; ++i) {
            float mx = fmaxf(c0[i], c1[i]);
#pragma unroll
            for (int o = 8; o >= 1; o >>= 1) mx = fmaxf(mx, __shfl_xor(mx, o, 32));
            const float mnew  = fmaxf(mrow[i], mx);
            const float alpha = __expf(mrow[i] - mnew);
            const float p0 = __expf(c0[i] - mnew);
            const float p1 = __expf(c1[i] - mnew);
            float rs = p0 + p1;
#pragma unroll
            for (int o = 8; o >= 1; o >>= 1) rs += __shfl_xor(rs, o, 32);
            lsum[i] = lsum[i] * alpha + rs;
            mrow[i] = mnew;
#pragma unroll
            for (int d = 0; d < 4; ++d) acc[d][i] *= alpha;
            Pbuf[widx][i + 8 * hl][ln]      = (_Float16)p0;
            Pbuf[widx][i + 8 * hl][16 + ln] = (_Float16)p1;
        }

        asm volatile("s_wait_dscnt 0" ::: "memory");  // wave-local LDS transpose
        const _Float16* prow = &Pbuf[widx][ln][0];
        const v16h ap = load_frag(prow + hl * 8, prow + 16 + hl * 8);

        // ---- PV: 4 hd tiles, K = 32 k-positions; batch the V fragments ----
        v16h vf[4];
#pragma unroll
        for (int d = 0; d < 4; ++d) {
            const _Float16* vr = &Vt[buf][d * 16 + ln][hl * 8];
            vf[d] = load_frag(vr, vr + 16);
        }
#pragma unroll
        for (int d = 0; d < 4; ++d)
            acc[d] = WMMA_F16(ap, vf[d], acc[d]);
    }

    // ---- epilogue: normalize and write ctx [B,S,D] fp32 ----
    const int b_ = bh >> 4, h = bh & 15;
#pragma unroll
    for (int d = 0; d < 4; ++d) {
#pragma unroll
        for (int i = 0; i < 8; ++i) {
            const int row = q0 + i + 8 * hl;
            ctx[((size_t)b_ * SS + row) * DD + h * 64 + d * 16 + ln] =
                acc[d][i] / lsum[i];
        }
    }
}

// ---------------------------------------------------------------------------
extern "C" void kernel_launch(void* const* d_in, const int* in_sizes, int n_in,
                              void* d_out, int out_size, void* d_ws, size_t ws_size,
                              hipStream_t stream) {
    const float* X  = (const float*)d_in[0];
    const float* Wq = (const float*)d_in[1];
    const float* bq = (const float*)d_in[2];
    const float* Wk = (const float*)d_in[3];
    const float* bk = (const float*)d_in[4];
    const float* Wv = (const float*)d_in[5];
    const float* bv = (const float*)d_in[6];

    char* ws = (char*)d_ws;
    _Float16* Xh  = (_Float16*)(ws + OFF_XH);
    _Float16* Wth = (_Float16*)(ws + OFF_WTH);
    _Float16* Qh  = (_Float16*)(ws + OFF_QH);
    _Float16* Kh  = (_Float16*)(ws + OFF_KH);
    _Float16* Vth = (_Float16*)(ws + OFF_VTH);

    float* ctx    = (float*)d_out;                        // [B,S,D]
    float* scores = (float*)d_out + (size_t)MROWS * DD;   // [B,H,S,S]

    // 0) fp32 -> f16 staging (X straight, W transposed)
    const int n4 = MROWS * DD / 4;                        // 2,097,152
    cvt_x_kernel<<<n4 / 256, 256, 0, stream>>>(X, Xh, n4);
    cvt_wt_kernel<<<(3 * 1024 * 1024) / 256, 256, 0, stream>>>(Wq, Wk, Wv, Wth);

    // 1) fused QKV projection + RoPE: 64 row-groups x 48 col-groups
    proj_rope_kernel<<<64 * 48, 256, 0, stream>>>(Xh, Wth, bq, bk, bv, Qh, Kh, Vth);

    // 2) attention: 64 (b,h) x 16 blocks, 8 q-tiles per block
    attn_kernel<<<64 * 16, 256, 0, stream>>>(Qh, Kh, Vth, ctx, scores);
}
